// Decoder_17781164605947
// MI455X (gfx1250) — matrix operand order repaired and checked by reading, not compiled, not run on hardware
//
#include <hip/hip_runtime.h>
#include <math.h>

// Input-feeding GRU decoder + Luong attention, 128 sequential steps.
// Single PERSISTENT kernel: 64 blocks x 512 threads loop over all steps with
// a device-wide generation barrier (the schedule is launch-latency bound:
// ~900 dependent kernel nodes -> ~10). GEMMs use v_wmma_f32_16x16x32_bf16 with
// register double-buffering; attention is a fused single-pass online softmax.

#define B_   64
#define TDEC 128
#define TENC 256
#define H_   1024
#define DX   1024

#define NBLK 64          // must all be co-resident (one WG per batch row in attn)
#define NTHR 512         // 16 waves per block
#define NWAVE (NBLK * (NTHR / 32))   // 1024 waves
#define NLANES (NBLK * NTHR)         // 32768 lanes

typedef __attribute__((ext_vector_type(16))) __bf16 v16bf;
typedef __attribute__((ext_vector_type(8)))  float  v8f;

static __device__ __forceinline__ __bf16 f2bf(float f) {
  union { float f; unsigned u; } a; a.f = f;
  unsigned r = (a.u + 0x7FFFu + ((a.u >> 16) & 1u)) >> 16;  // RNE
  union { unsigned short s; __bf16 b; } o; o.s = (unsigned short)r;
  return o.b;
}

// A-operand packed layout: [Mtiles][KtTot][32 lanes][16 bf16]
static __device__ __forceinline__ void storePackedA(__bf16* dst, int KtTot, int m, int k, float v) {
  int kt = k >> 5, ko = k & 31;
  int khalf = (ko >> 3) & 1;
  int j = (ko & 7) | ((ko >> 4) << 3);
  int lane = (khalf << 4) | (m & 15);
  int mt = m >> 4;
  dst[((size_t)(mt * KtTot + kt) * 32 + lane) * 16 + j] = f2bf(v);
}

// B-operand packed layout: [Ktiles][Ntiles][32 lanes][16 bf16]
static __device__ __forceinline__ void storePackedB(__bf16* dst, int Ntiles, int k, int n, float v) {
  int kt = k >> 5, ko = k & 31;
  int khalf = (ko >> 3) & 1;
  int j = (ko & 7) | ((ko >> 4) << 3);
  int lane = (khalf << 4) | (n & 15);
  int nt = n >> 4;
  dst[((size_t)(kt * Ntiles + nt) * 32 + lane) * 16 + j] = f2bf(v);
}

__global__ void pack_w_kernel(const float* __restrict__ W, __bf16* __restrict__ dst, int K, int N) {
  int i = blockIdx.x * blockDim.x + threadIdx.x;
  if (i >= K * N) return;
  int k = i / N, n = i - k * N;
  storePackedB(dst, N >> 4, k, n, W[i]);
}

__global__ void pack_a_kernel(const float* __restrict__ src, int rowStride, int Kcols, int kBase,
                              __bf16* __restrict__ dst, int KtTot) {
  int i = blockIdx.x * blockDim.x + threadIdx.x;
  if (i >= B_ * Kcols) return;
  int m = i / Kcols, k = i - m * Kcols;
  storePackedA(dst, KtTot, m, kBase + k, src[(size_t)m * rowStride + k]);
}

// ---- device-wide barrier (all NBLK blocks resident) ----
static __device__ __forceinline__ void grid_barrier(int* cnt, int* gen) {
  __threadfence();          // make this thread's writes device-visible
  __syncthreads();          // whole block has fenced
  if (threadIdx.x == 0) {
    int g = __hip_atomic_load(gen, __ATOMIC_RELAXED, __HIP_MEMORY_SCOPE_AGENT);
    if (atomicAdd(cnt, 1) == NBLK - 1) {
      __hip_atomic_store(cnt, 0, __ATOMIC_RELAXED, __HIP_MEMORY_SCOPE_AGENT);
      __hip_atomic_store(gen, g + 1, __ATOMIC_RELEASE, __HIP_MEMORY_SCOPE_AGENT);
    } else {
      while (__hip_atomic_load(gen, __ATOMIC_ACQUIRE, __HIP_MEMORY_SCOPE_AGENT) <= g)
        __builtin_amdgcn_s_sleep(1);
    }
  }
  __syncthreads();
  __threadfence();          // acquire: discard stale cached data
}

// One 16 x (NT*16) output strip per wave; register double-buffered k-loop.
template <int NT>
static __device__ __forceinline__ void gemm_strip(
    int strip, int lane,
    const v16bf* __restrict__ Ap, const v16bf* __restrict__ Bp,
    const float* __restrict__ bias,
    float* __restrict__ outF, int ldOut,
    __bf16* __restrict__ outP, int outPKtTot, int outPkBase,
    int Ktiles, int Ntiles, int act) {
  int groups = Ntiles / NT;
  int mt  = strip / groups;
  int nt0 = (strip - mt * groups) * NT;

  v8f zero = {0.f, 0.f, 0.f, 0.f, 0.f, 0.f, 0.f, 0.f};
  v8f acc[NT];
#pragma unroll
  for (int j = 0; j < NT; ++j) acc[j] = zero;

  const v16bf* aP = Ap + (size_t)mt * Ktiles * 32 + lane;
  v16bf a = aP[0];
  v16bf b[NT];
  {
    const v16bf* bP = Bp + (size_t)nt0 * 32 + lane;
#pragma unroll
    for (int j = 0; j < NT; ++j) b[j] = bP[(size_t)j * 32];
  }
  for (int kt = 0; kt < Ktiles - 1; ++kt) {
    v16bf aN = aP[(size_t)(kt + 1) * 32];
    const v16bf* bN = Bp + ((size_t)(kt + 1) * Ntiles + nt0) * 32 + lane;
    v16bf c[NT];
#pragma unroll
    for (int j = 0; j < NT; ++j) c[j] = bN[(size_t)j * 32];
#pragma unroll
    for (int j = 0; j < NT; ++j)
      acc[j] = __builtin_amdgcn_wmma_f32_16x16x32_bf16(false, a, false, b[j], (short)0, acc[j], false, false);
    a = aN;
#pragma unroll
    for (int j = 0; j < NT; ++j) b[j] = c[j];
  }
#pragma unroll
  for (int j = 0; j < NT; ++j)
    acc[j] = __builtin_amdgcn_wmma_f32_16x16x32_bf16(false, a, false, b[j], (short)0, acc[j], false, false);

  int mrow0 = mt * 16 + ((lane >> 4) << 3);
#pragma unroll
  for (int j = 0; j < NT; ++j) {
    int n = (nt0 + j) * 16 + (lane & 15);
    float bv = bias[n];
#pragma unroll
    for (int v = 0; v < 8; ++v) {
      int m = mrow0 + v;
      float val = acc[j][v] + bv;
      if (act) val = tanhf(val);
      if (outF) outF[(size_t)m * ldOut + n] = val;
      if (outP) storePackedA(outP, outPKtTot, m, outPkBase + n, val);
    }
  }
}

// Fused scores+softmax+value (per block = one batch row), online softmax.
static __device__ __forceinline__ void attn_phase(int b, int lane, int wid,
                                                  const float* __restrict__ h,
                                                  const float* __restrict__ o_enc,
                                                  __bf16* __restrict__ cat2P) {
  __shared__ float sm_m[16];
  __shared__ float sm_l[16];
  __shared__ float sval[H_];

  float4 hreg[8];
  const float4* hp = (const float4*)(h + (size_t)b * H_);
#pragma unroll
  for (int i = 0; i < 8; ++i) hreg[i] = hp[lane + 32 * i];

  float4 acc[8];
#pragma unroll
  for (int i = 0; i < 8; ++i) acc[i] = make_float4(0.f, 0.f, 0.f, 0.f);
  float m = -INFINITY, l = 0.f;

  for (int te = wid; te < TENC; te += 16) {
    const float4* row = (const float4*)(o_enc + ((size_t)b * TENC + te) * H_);
    float4 r[8];
    float s = 0.f;
#pragma unroll
    for (int i = 0; i < 8; ++i) {
      r[i] = row[lane + 32 * i];
      s += hreg[i].x * r[i].x + hreg[i].y * r[i].y + hreg[i].z * r[i].z + hreg[i].w * r[i].w;
    }
    for (int off = 16; off; off >>= 1) s += __shfl_xor(s, off, 32);
    float mN = fmaxf(m, s);
    float sc = expf(m - mN);
    float p  = expf(s - mN);
#pragma unroll
    for (int i = 0; i < 8; ++i) {
      acc[i].x = acc[i].x * sc + r[i].x * p;
      acc[i].y = acc[i].y * sc + r[i].y * p;
      acc[i].z = acc[i].z * sc + r[i].z * p;
      acc[i].w = acc[i].w * sc + r[i].w * p;
    }
    l = l * sc + p;
    m = mN;
  }

  if (lane == 0) { sm_m[wid] = m; sm_l[wid] = l; }
  for (int idx = threadIdx.x; idx < H_; idx += NTHR) sval[idx] = 0.f;
  __syncthreads();
  float M = -INFINITY;
#pragma unroll
  for (int w = 0; w < 16; ++w) M = fmaxf(M, sm_m[w]);
  float L = 0.f;
#pragma unroll
  for (int w = 0; w < 16; ++w) L += sm_l[w] * expf(sm_m[w] - M);
  float factor = expf(m - M);
#pragma unroll
  for (int i = 0; i < 8; ++i) {
    int j = 4 * lane + 128 * i;
    atomicAdd(&sval[j + 0], acc[i].x * factor);
    atomicAdd(&sval[j + 1], acc[i].y * factor);
    atomicAdd(&sval[j + 2], acc[i].z * factor);
    atomicAdd(&sval[j + 3], acc[i].w * factor);
  }
  __syncthreads();
  float invL = 1.f / L;
  for (int idx = threadIdx.x; idx < H_; idx += NTHR)
    storePackedA(cat2P, 64, b, idx, sval[idx] * invL);
}

__global__ void __launch_bounds__(NTHR) decoder_persistent(
    const float* __restrict__ x, const float* __restrict__ o_enc,
    const float* __restrict__ bfeed, const float* __restrict__ bxi,
    const float* __restrict__ bhr, const float* __restrict__ batt,
    const v16bf* __restrict__ WfeedP, const v16bf* __restrict__ WxP,
    const v16bf* __restrict__ WhP, const v16bf* __restrict__ WattP,
    __bf16* ifeedP, __bf16* icatP, __bf16* hP, __bf16* cat2P,
    float* hF, float* gxB, float* ghB,
    float* out, int* barCnt, int* barGen) {
  int lane = threadIdx.x & 31;
  int wid  = threadIdx.x >> 5;
  int gw   = blockIdx.x * (NTHR / 32) + wid;   // 0..1023
  int tid  = blockIdx.x * NTHR + threadIdx.x;  // 0..32767

  for (int t = 0; t < TDEC; ++t) {
    // ---- Phase A (independent): feed-GEMM | gh-GEMM | pack x_t ----
    if (gw < 128) {
      // tanh(i_feed @ Wfeed + bfeed) -> icat[:, 0:1024] (packed A)
      gemm_strip<2>(gw, lane, (const v16bf*)ifeedP, WfeedP, bfeed,
                    (float*)nullptr, 0, icatP, 64, 0, 32, 64, 1);
    } else if (gw < 512) {
      // gh = h @ Wh + bhr
      gemm_strip<2>(gw - 128, lane, (const v16bf*)hP, WhP, bhr,
                    ghB, 3 * H_, (__bf16*)nullptr, 0, 0, 32, 192, 0);
    } else {
      // x_t -> icat[:, 1024:2048] (packed A)
      int p = (gw - 512) * 32 + lane;
      for (int idx = p; idx < B_ * DX; idx += 512 * 32) {
        int m = idx >> 10, k = idx & 1023;
        storePackedA(icatP, 64, m, 1024 + k, x[((size_t)m * TDEC + t) * DX + k]);
      }
    }
    grid_barrier(barCnt, barGen);

    // ---- Phase B: gx = icat @ Wx + bxi (768 waves, 1 N-tile each) ----
    if (gw < 768) {
      gemm_strip<1>(gw, lane, (const v16bf*)icatP, WxP, bxi,
                    gxB, 3 * H_, (__bf16*)nullptr, 0, 0, 64, 192, 0);
    }
    grid_barrier(barCnt, barGen);

    // ---- Phase C: GRU gates -> h_new ----
    for (int i = tid; i < B_ * H_; i += NLANES) {
      int b = i >> 10, j = i & 1023;
      const float* gxr = gxB + (size_t)b * 3 * H_;
      const float* ghr = ghB + (size_t)b * 3 * H_;
      float xz = gxr[j], xr = gxr[H_ + j], xh = gxr[2 * H_ + j];
      float hz = ghr[j], hr = ghr[H_ + j], hh = ghr[2 * H_ + j];
      float hp = hF[i];
      float z  = 1.f / (1.f + expf(-(xz + hz)));
      float r  = 1.f / (1.f + expf(-(xr + hr)));
      float hc = tanhf(xh + r * hh);
      float hn = z * hp + (1.f - z) * hc;
      hF[i] = hn;
      storePackedA(hP, 32, b, j, hn);           // next-step recurrent GEMM input
      storePackedA(cat2P, 64, b, H_ + j, hn);   // upper half of [value, h_new]
    }
    grid_barrier(barCnt, barGen);

    // ---- Phase D: fused attention (block = batch row, o_enc read once) ----
    attn_phase(blockIdx.x, lane, wid, hF, o_enc, cat2P);
    grid_barrier(barCnt, barGen);

    // ---- Phase E: h_att = tanh([value,h_new] @ Watt + batt) ----
    if (gw < 128) {
      gemm_strip<2>(gw, lane, (const v16bf*)cat2P, WattP, batt,
                    out + (size_t)t * H_, TDEC * H_, ifeedP, 32, 0, 64, 64, 1);
    }
    grid_barrier(barCnt, barGen);
  }
}

extern "C" void kernel_launch(void* const* d_in, const int* in_sizes, int n_in,
                              void* d_out, int out_size, void* d_ws, size_t ws_size,
                              hipStream_t stream) {
  (void)in_sizes; (void)n_in; (void)out_size; (void)ws_size;
  const float* x     = (const float*)d_in[0];
  const float* o_enc = (const float*)d_in[1];
  const float* h_enc = (const float*)d_in[2];
  const float* Wfeed = (const float*)d_in[3];
  const float* bfeed = (const float*)d_in[4];
  const float* Wx    = (const float*)d_in[5];
  const float* Wh    = (const float*)d_in[6];
  const float* bxi   = (const float*)d_in[7];
  const float* bhr   = (const float*)d_in[8];
  const float* Watt  = (const float*)d_in[9];
  const float* batt  = (const float*)d_in[10];
  float* out = (float*)d_out;

  char* ws = (char*)d_ws;
  size_t off = 0;
  auto alloc = [&](size_t bytes) { size_t r = off; off = (off + bytes + 255) & ~(size_t)255; return r; };
  __bf16* WfeedP = (__bf16*)(ws + alloc((size_t)32 * 64  * 512 * 2));
  __bf16* WxP    = (__bf16*)(ws + alloc((size_t)64 * 192 * 512 * 2));
  __bf16* WhP    = (__bf16*)(ws + alloc((size_t)32 * 192 * 512 * 2));
  __bf16* WattP  = (__bf16*)(ws + alloc((size_t)64 * 64  * 512 * 2));
  __bf16* ifeedP = (__bf16*)(ws + alloc((size_t)4 * 32 * 512 * 2));
  __bf16* icatP  = (__bf16*)(ws + alloc((size_t)4 * 64 * 512 * 2));
  __bf16* hP     = (__bf16*)(ws + alloc((size_t)4 * 32 * 512 * 2));
  __bf16* cat2P  = (__bf16*)(ws + alloc((size_t)4 * 64 * 512 * 2));
  float*  hF     = (float*)(ws + alloc((size_t)B_ * H_ * 4));
  float*  gxB    = (float*)(ws + alloc((size_t)B_ * 3 * H_ * 4));
  float*  ghB    = (float*)(ws + alloc((size_t)B_ * 3 * H_ * 4));
  int*    bar    = (int*)(ws + alloc(256));   // [0]=count, [1]=generation
  // total ~26 MB of scratch

  // ---- prep: pack weights to WMMA B layout (bf16), init state ----
  pack_w_kernel<<<(1024 * 1024 + 255) / 256, 256, 0, stream>>>(Wfeed, WfeedP, 1024, 1024);
  pack_w_kernel<<<(2048 * 3072 + 255) / 256, 256, 0, stream>>>(Wx,    WxP,    2048, 3072);
  pack_w_kernel<<<(1024 * 3072 + 255) / 256, 256, 0, stream>>>(Wh,    WhP,    1024, 3072);
  pack_w_kernel<<<(2048 * 1024 + 255) / 256, 256, 0, stream>>>(Watt,  WattP,  2048, 1024);
  hipMemcpyAsync(hF, h_enc, (size_t)B_ * H_ * 4, hipMemcpyDeviceToDevice, stream);
  pack_a_kernel<<<(B_ * H_ + 255) / 256, 256, 0, stream>>>(h_enc, H_, H_, 0, hP, 32);
  hipMemsetAsync(ifeedP, 0, (size_t)4 * 32 * 512 * 2, stream);   // i_feed0 = 0
  hipMemsetAsync(bar, 0, 256, stream);                           // barrier state

  // ---- one persistent kernel runs all 128 steps ----
  decoder_persistent<<<NBLK, NTHR, 0, stream>>>(
      x, o_enc, bfeed, bxi, bhr, batt,
      (const v16bf*)WfeedP, (const v16bf*)WxP, (const v16bf*)WhP, (const v16bf*)WattP,
      ifeedP, icatP, hP, cat2P, hF, gxB, ghB, out, bar, bar + 1);
}
